// ChatGPTAttention_55533927137517
// MI455X (gfx1250) — matrix operand order repaired and checked by reading, not compiled, not run on hardware
//
// MI455X (gfx1250) multi-head causal attention, bf16 WMMA throughout.
// GEMMs use the Tensor Data Mover (tensor_load_to_lds, 6-arg builtin form) to
// stage double-buffered LDS tiles (s_wait_tensorcnt sync), wave tile 2x4 WMMA.
// Flash-attention (S^T = K*Q^T formulation) unchanged.
#include <hip/hip_runtime.h>

#define DMODEL 1024
#define NHEADS 16
#define DKH    64
#define BATCH  2
#define SEQ    2048
#define NTOK   (BATCH * SEQ)

#define STAGE_K    32              // K-depth staged per LDS buffer
#define LDS_STRIDE 40              // 32 elems + 8 pad -> 80B rows, bank-conflict-free

typedef __bf16 bf16;
typedef __attribute__((ext_vector_type(16))) __bf16 bf16x16;
typedef __attribute__((ext_vector_type(8)))  float  f32x8;
typedef __attribute__((ext_vector_type(4)))  unsigned int u32x4;
typedef __attribute__((ext_vector_type(8)))  int          i32x8;
typedef __attribute__((ext_vector_type(4)))  int          i32x4;

union Frag16 { bf16x16 v; uint4 u[2]; };
union Pack8  { uint4 u;   bf16 e[8];  };

#if defined(__has_builtin)
#if __has_builtin(__builtin_amdgcn_tensor_load_to_lds) && \
    __has_builtin(__builtin_amdgcn_s_wait_tensorcnt)
#define HAVE_TDM 1
#endif
#endif
#ifndef HAVE_TDM
#define HAVE_TDM 0
#endif

// A-matrix fragment (16x32 MxK, bf16): lane L holds row M = L%16.
// lanes 0-15: K = k0..k0+15; lanes 16-31: +16.
static __device__ __forceinline__ bf16x16
load_a_frag(const bf16* __restrict__ base, int ld, int row, int k0, int lane) {
  const int hi = (lane >> 4) & 1;
  const bf16* p = base + (size_t)row * (size_t)ld + k0 + 16 * hi;
  Frag16 f;
  f.u[0] = *(const uint4*)(p);
  f.u[1] = *(const uint4*)(p + 8);
  return f.v;
}

// B-matrix fragment (32x16 KxN, bf16), loaded from B^T rows (row-major, K inner):
// lane L holds column N = L%16; lanes 0-15: K = k0..k0+15, lanes 16-31: +16.
static __device__ __forceinline__ bf16x16
load_b_frag(const bf16* __restrict__ base, int ld, int row, int k0, int lane) {
  const int hi = (lane >> 4) & 1;
  const bf16* p = base + (size_t)row * (size_t)ld + k0 + 16 * hi;
  Frag16 f;
  f.u[0] = *(const uint4*)(p);
  f.u[1] = *(const uint4*)(p + 8);
  return f.v;
}

static __device__ __forceinline__ f32x8 wmma_bf16(bf16x16 a, bf16x16 b, f32x8 c) {
  return __builtin_amdgcn_wmma_f32_16x16x32_bf16(false, a, false, b, (short)0, c,
                                                 false, false);
}

#if HAVE_TDM
// TDM: 2D tile load, rows x 32 bf16 elements, row stride = row_stride elems in
// global, LDS rows padded 64B->80B (pad_interval code 3 = 8x8B, pad_amount
// code 3 = 4 DWORDs).  D# layout per CDNA5 ISA §8.3/8.4.
static __device__ __forceinline__ void
tdm_load_tile32(unsigned lds_addr, const bf16* gaddr, int tile_rows, int row_stride) {
  unsigned long long ga = (unsigned long long)(uintptr_t)gaddr;
  u32x4 g0;
  g0[0] = 1u;                                      // count=1, user desc, no gather
  g0[1] = lds_addr;                                // LDS byte address
  g0[2] = (unsigned)(ga & 0xffffffffu);            // global_addr[31:0]
  g0[3] = (unsigned)((ga >> 32) & 0x1ffffffu) | (2u << 30);  // addr[56:32], type=2
  i32x8 g1;
  g1[0] = (1 << 16) |                              // data_size = 2 bytes
          (1 << 20) |                              // pad_enable
          (3 << 22) |                              // pad_interval: 8 x 8B = 64B
          (3 << 25);                               // pad_amount: 4 DWORDs = 16B
  g1[1] = (row_stride & 0xffff) << 16;             // tensor_dim0[15:0]
  g1[2] = ((row_stride >> 16) & 0xffff) | ((tile_rows & 0xffff) << 16); // dim0 hi, dim1 lo
  g1[3] = (STAGE_K & 0xffff) << 16;                // tile_dim0 = 32
  g1[4] = tile_rows & 0xffff;                      // tile_dim1; tile_dim2 = 0
  g1[5] = row_stride;                              // tensor_dim0_stride[31:0]
  g1[6] = 0;                                       // stride hi / dim1_stride
  g1[7] = 0;
  i32x4 z4 = {0, 0, 0, 0};                         // groups 2/3 unused (2D tensor)
  i32x8 z8 = {0, 0, 0, 0, 0, 0, 0, 0};             // extra group (6-arg form)
  __builtin_amdgcn_tensor_load_to_lds(g0, g1, z4, z4, z8, 0);
}
#else
// Fallback: cooperative copy of rows x 32 bf16 into padded LDS tile.
static __device__ __forceinline__ void
copy_tile32(bf16* lds, const bf16* g, int rows, int row_stride, int tid) {
  for (int i = tid; i < rows * 4; i += 256) {
    int r = i >> 2, c = i & 3;
    *(uint4*)(lds + r * LDS_STRIDE + c * 8) =
        *(const uint4*)(g + (size_t)r * row_stride + c * 8);
  }
}
#endif

__global__ void cvt_f32_bf16(const float* __restrict__ src, bf16* __restrict__ dst, int n) {
  int i = blockIdx.x * blockDim.x + threadIdx.x;
  int stride = gridDim.x * blockDim.x;
  for (; i < n; i += stride) dst[i] = (bf16)src[i];
}

// Shared GEMM body: block tile 64 tokens x 256 features, 8 waves, each wave a
// 32x64 tile (2x4 WMMA).  A/B tiles TDM-staged in double-buffered LDS.
// EMIT = 0: scatter bf16 to Q/K/Vt ; EMIT = 1: fp32 out with bias.
template <int EMIT>
static __device__ __forceinline__ void gemm_body(
    const bf16* __restrict__ X, const bf16* __restrict__ W,
    const float* __restrict__ bias,
    bf16* __restrict__ Qd, bf16* __restrict__ Kd, bf16* __restrict__ Vt,
    float* __restrict__ out) {
  __shared__ __attribute__((aligned(16))) bf16 As[2][64 * LDS_STRIDE];
  __shared__ __attribute__((aligned(16))) bf16 Bs[2][256 * LDS_STRIDE];

  const int lane = threadIdx.x & 31;
  const int wave = threadIdx.x >> 5;
  const int lr = lane & 15;
  const int hi = lane >> 4;
  const int rowA = (wave & 1) * 32;   // wave's token rows within the 64 tile
  const int rowB = (wave >> 1) * 64;  // wave's feature rows within the 256 tile
  const int t0 = blockIdx.x * 64;
  const int f0 = blockIdx.y * 256;

  const bf16* Abase = X + (size_t)t0 * DMODEL;
  const bf16* Bbase = W + (size_t)f0 * DMODEL;

  f32x8 acc[2][4] = {};
  const int NSTAGE = DMODEL / STAGE_K;  // 32

#if HAVE_TDM
  if (wave == 0) {
    tdm_load_tile32((unsigned)(uintptr_t)&As[0][0], Abase, 64, DMODEL);
    tdm_load_tile32((unsigned)(uintptr_t)&Bs[0][0], Bbase, 256, DMODEL);
  }
#endif
  for (int s = 0; s < NSTAGE; ++s) {
    const int buf = s & 1;
    __syncthreads();  // readers finished with buf^1 (its last use was stage s-1)
#if HAVE_TDM
    if (wave == 0) {
      if (s + 1 < NSTAGE) {
        tdm_load_tile32((unsigned)(uintptr_t)&As[buf ^ 1][0],
                        Abase + (s + 1) * STAGE_K, 64, DMODEL);
        tdm_load_tile32((unsigned)(uintptr_t)&Bs[buf ^ 1][0],
                        Bbase + (s + 1) * STAGE_K, 256, DMODEL);
        // TDM ops complete in order per wave: <=2 outstanding means the 2 ops
        // that filled buf (current) are done.
        __builtin_amdgcn_s_wait_tensorcnt((short)2);
      } else {
        __builtin_amdgcn_s_wait_tensorcnt((short)0);
      }
    }
#else
    copy_tile32(&As[buf][0], Abase + s * STAGE_K, 64, DMODEL, threadIdx.x);
    copy_tile32(&Bs[buf][0], Bbase + s * STAGE_K, 256, DMODEL, threadIdx.x);
#endif
    __syncthreads();  // buf is ready for all waves

    bf16x16 a0 = load_a_frag(&As[buf][0], LDS_STRIDE, rowA + lr,      0, lane);
    bf16x16 a1 = load_a_frag(&As[buf][0], LDS_STRIDE, rowA + 16 + lr, 0, lane);
#pragma unroll
    for (int j = 0; j < 4; ++j) {
      bf16x16 bj = load_b_frag(&Bs[buf][0], LDS_STRIDE, rowB + j * 16 + lr, 0, lane);
      acc[0][j] = wmma_bf16(a0, bj, acc[0][j]);
      acc[1][j] = wmma_bf16(a1, bj, acc[1][j]);
    }
  }

  // C-layout: lane -> col n0 + L%16; VGPR v -> row m0 + v + 8*(L/16).
#pragma unroll
  for (int i = 0; i < 2; ++i)
#pragma unroll
    for (int j = 0; j < 4; ++j)
#pragma unroll
      for (int v = 0; v < 8; ++v) {
        int t = t0 + rowA + i * 16 + v + 8 * hi;
        int f = f0 + rowB + j * 16 + lr;
        float val = acc[i][j][v] + bias[f];
        if (EMIT == 1) {
          out[(size_t)t * DMODEL + f] = val;
        } else {
          int bb = t >> 11, sIdx = t & (SEQ - 1);
          int sel = f >> 10, fm = f & (DMODEL - 1);
          int h = fm >> 6, dk = fm & 63;
          size_t bh = (size_t)(bb * NHEADS + h);
          if (sel == 0)      Qd[(bh * SEQ + sIdx) * DKH + dk] = (bf16)val;
          else if (sel == 1) Kd[(bh * SEQ + sIdx) * DKH + dk] = (bf16)val;
          else               Vt[(bh * DKH + dk) * SEQ + sIdx] = (bf16)val;
        }
      }
}

__global__ __launch_bounds__(256) void qkv_gemm(
    const bf16* __restrict__ X, const bf16* __restrict__ W,
    const float* __restrict__ bias,
    bf16* __restrict__ Qd, bf16* __restrict__ Kd, bf16* __restrict__ Vt) {
  gemm_body<0>(X, W, bias, Qd, Kd, Vt, nullptr);
}

__global__ __launch_bounds__(256) void out_proj(
    const bf16* __restrict__ A, const bf16* __restrict__ W,
    const float* __restrict__ bias, float* __restrict__ out) {
  gemm_body<1>(A, W, bias, nullptr, nullptr, nullptr, out);
}

// Flash attention, one wave per (b, h, 16-query tile).
// S^T = K * Q^T so each lane owns one query column -> softmax stats are
// 16 in-lane values + one shfl_xor(16).  O^T accumulated as V^T * P^T.
__global__ __launch_bounds__(256) void attn_fa(
    const bf16* __restrict__ Q, const bf16* __restrict__ K,
    const bf16* __restrict__ Vt, bf16* __restrict__ AO) {
  const int lane = threadIdx.x & 31;
  const int lr = lane & 15;
  const int hi = lane >> 4;
  const int wid = blockIdx.x * 8 + (threadIdx.x >> 5);
  const int qt = wid & (SEQ / 16 - 1);
  const int h  = (wid >> 7) & (NHEADS - 1);
  const int b  = wid >> 11;

  const size_t bh = (size_t)(b * NHEADS + h);
  const bf16* Qp = Q  + bh * SEQ * DKH;
  const bf16* Kp = K  + bh * SEQ * DKH;
  const bf16* Vp = Vt + bh * DKH * SEQ;

  const int q0 = qt * 16;
  const int q  = q0 + lr;  // this lane's query index

  bf16x16 qb0 = load_b_frag(Qp, DKH, q0 + lr, 0,  lane);
  bf16x16 qb1 = load_b_frag(Qp, DKH, q0 + lr, 32, lane);

  f32x8 o[4] = {};
  float m_run = -1e30f, l_run = 0.0f;
  const float scale = 0.125f * 1.44269504088896f;  // 1/sqrt(64) * log2(e)

  const int kend = q0 + 16;
  for (int kb = 0; kb < kend; kb += 32) {
    f32x8 s0 = {}, s1 = {};
    {
      bf16x16 a00 = load_a_frag(Kp, DKH, kb + lr,      0,  lane);
      bf16x16 a01 = load_a_frag(Kp, DKH, kb + lr,      32, lane);
      bf16x16 a10 = load_a_frag(Kp, DKH, kb + 16 + lr, 0,  lane);
      bf16x16 a11 = load_a_frag(Kp, DKH, kb + 16 + lr, 32, lane);
      s0 = wmma_bf16(a00, qb0, s0);
      s0 = wmma_bf16(a01, qb1, s0);
      s1 = wmma_bf16(a10, qb0, s1);
      s1 = wmma_bf16(a11, qb1, s1);
    }
    float sv[16];
#pragma unroll
    for (int v = 0; v < 8; ++v) {
      int k0i = kb + 8 * hi + v;
      int k1i = kb + 16 + 8 * hi + v;
      sv[v]     = (k0i <= q) ? s0[v] * scale : -1e30f;
      sv[8 + v] = (k1i <= q) ? s1[v] * scale : -1e30f;
    }
    float mb = -1e30f;
#pragma unroll
    for (int i = 0; i < 16; ++i) mb = fmaxf(mb, sv[i]);
    mb = fmaxf(mb, __shfl_xor(mb, 16));
    float m_new = fmaxf(m_run, mb);
    float alpha = exp2f(m_run - m_new);
    float p[16];
    float ls = 0.0f;
#pragma unroll
    for (int i = 0; i < 16; ++i) { p[i] = exp2f(sv[i] - m_new); ls += p[i]; }
    ls += __shfl_xor(ls, 16);
    l_run = l_run * alpha + ls;
    m_run = m_new;
#pragma unroll
    for (int t = 0; t < 4; ++t)
#pragma unroll
      for (int v = 0; v < 8; ++v) o[t][v] = o[t][v] * alpha;

    // Repack P into B-fragment layout: lanes 0-15 need keys kb+0..15,
    // lanes 16-31 need kb+16..31 -> single shfl_xor(16) swap.
    bf16x16 pf;
#pragma unroll
    for (int v = 0; v < 8; ++v) {
      float sendv = hi ? p[v] : p[8 + v];
      float recv  = __shfl_xor(sendv, 16);
      float e0 = hi ? recv     : p[v];
      float e1 = hi ? p[8 + v] : recv;
      pf[v]     = (bf16)e0;
      pf[8 + v] = (bf16)e1;
    }
#pragma unroll
    for (int t = 0; t < 4; ++t) {
      bf16x16 va = load_a_frag(Vp, SEQ, t * 16 + lr, kb, lane);
      o[t] = wmma_bf16(va, pf, o[t]);
    }
  }

  const float inv = 1.0f / l_run;
  size_t obase = ((size_t)(b * SEQ + q)) * DMODEL + h * DKH;
#pragma unroll
  for (int t = 0; t < 4; ++t) {
    Pack8 pk;
#pragma unroll
    for (int v = 0; v < 8; ++v) pk.e[v] = (bf16)(o[t][v] * inv);
    *(uint4*)(AO + obase + t * 16 + 8 * hi) = pk.u;
  }
}

extern "C" void kernel_launch(void* const* d_in, const int* in_sizes, int n_in,
                              void* d_out, int out_size, void* d_ws, size_t ws_size,
                              hipStream_t stream) {
  (void)in_sizes; (void)n_in; (void)out_size; (void)ws_size;
  const float* x    = (const float*)d_in[0];
  // d_in[1] = causal mask, implemented analytically in attn_fa
  const float* wqkv = (const float*)d_in[2];
  const float* bqkv = (const float*)d_in[3];
  const float* wo   = (const float*)d_in[4];
  const float* bo   = (const float*)d_in[5];
  float* out = (float*)d_out;

  bf16* p = (bf16*)d_ws;
  bf16* Xb  = p; p += (size_t)NTOK * DMODEL;
  bf16* Wqb = p; p += (size_t)3 * DMODEL * DMODEL;
  bf16* Wob = p; p += (size_t)DMODEL * DMODEL;
  bf16* Qd  = p; p += (size_t)NTOK * DMODEL;
  bf16* Kd  = p; p += (size_t)NTOK * DMODEL;
  bf16* Vt  = p; p += (size_t)NTOK * DMODEL;
  bf16* AO  = p; p += (size_t)NTOK * DMODEL;   // total 48 MiB

  cvt_f32_bf16<<<2048, 256, 0, stream>>>(x,    Xb,  NTOK * DMODEL);
  cvt_f32_bf16<<<2048, 256, 0, stream>>>(wqkv, Wqb, 3 * DMODEL * DMODEL);
  cvt_f32_bf16<<<1024, 256, 0, stream>>>(wo,   Wob, DMODEL * DMODEL);

  qkv_gemm<<<dim3(NTOK / 64, 3 * DMODEL / 256), 256, 0, stream>>>(
      Xb, Wqb, bqkv, Qd, Kd, Vt);

  attn_fa<<<(BATCH * NHEADS * (SEQ / 16)) / 8, 256, 0, stream>>>(Qd, Kd, Vt, AO);

  out_proj<<<dim3(NTOK / 64, DMODEL / 256), 256, 0, stream>>>(AO, Wob, bo, out);
}
